// MultiHeadSelfAttentionWithEdge_31052613550223
// MI455X (gfx1250) — compile-verified, matrix operand order checked by reading, not run on hardware
//
#include <hip/hip_runtime.h>
#include <math.h>

// ---------------------------------------------------------------------------
// MultiHeadSelfAttentionWithEdge for gfx1250 (MI455X)
// B=4, N=512, D_EMB=64, H=8, dk=8
// Memory-bound: stream e (256MB) once (NT loads), write e_out (256MB) once
// (NT stores).  WMMA (v_wmma_f32_16x16x32_f16) used for:
//   (a) e @ [W_e | W_g]  (K=64 -> two chained 16x16x32 WMMAs)
//   (b) _E^T @ O_e       (K=8 padded to 32, 4 WMMAs per 16x64 tile)
// ---------------------------------------------------------------------------

typedef __attribute__((ext_vector_type(16))) _Float16 v16h;
typedef __attribute__((ext_vector_type(8)))  float    v8f;
typedef __attribute__((ext_vector_type(4)))  float    v4f;

#define BB 4
#define NN 512
#define DE 64
#define NH 8
#define DK 8
#define LDSW 9   // padded row stride for lds_E (avoids 8-stride bank conflicts)

// -------------------------- Kernel 1: QKV projection ------------------------
// qkv[b,n,3,h,k] = n[b,n,:] @ W_qkv  -> scatter into Q/K/V as [B,H,N,dk] f32
__global__ void __launch_bounds__(192)
qkv_proj_kernel(const float* __restrict__ n_in, const float* __restrict__ Wqkv,
                float* __restrict__ Qb, float* __restrict__ Kb, float* __restrict__ Vb)
{
    const int row = blockIdx.x;             // b*512 + n
    const int b = row >> 9, nn = row & 511;
    __shared__ float nl[DE];
    const int t = threadIdx.x;              // 0..191 (output column)
    if (t < DE) nl[t] = n_in[(size_t)row * DE + t];
    __syncthreads();
    float acc = 0.f;
#pragma unroll
    for (int k = 0; k < DE; ++k) acc += nl[k] * Wqkv[k * 192 + t];
    const int which = t / 64, rem = t % 64, h = rem >> 3, kk = rem & 7;
    float* dst = (which == 0) ? Qb : (which == 1) ? Kb : Vb;
    dst[(((size_t)(b * NH + h) * NN) + nn) * DK + kk] = acc;
}

// --------------------- helpers for WMMA fragment packing --------------------
__device__ inline v16h pack_e_frag_nt(const float* lo, const float* hi)
{
    // elems 0..7 <- lo[0..7], elems 8..15 <- hi[0..7], f32 -> f16, NT b128 loads
    v4f l0 = __builtin_nontemporal_load((const v4f*)lo);
    v4f l1 = __builtin_nontemporal_load((const v4f*)(lo + 4));
    v4f h0 = __builtin_nontemporal_load((const v4f*)hi);
    v4f h1 = __builtin_nontemporal_load((const v4f*)(hi + 4));
    v16h a;
#pragma unroll
    for (int i = 0; i < 4; ++i) {
        a[i]      = (_Float16)l0[i];
        a[i + 4]  = (_Float16)l1[i];
        a[i + 8]  = (_Float16)h0[i];
        a[i + 12] = (_Float16)h1[i];
    }
    return a;
}

__device__ inline float fast_sigmoid(float x)
{
    // v_rcp_f32 instead of IEEE div chain: gate feeds log1p, 1-ulp rcp is fine
    return __builtin_amdgcn_rcpf(1.f + __expf(-x));
}

// ------------------- Kernel 2: fused attention + edge out -------------------
// One workgroup (128 threads = 4 waves) per (b, n) row.
__global__ void __launch_bounds__(128)
attn_edge_kernel(const float* __restrict__ e_in,
                 const float* __restrict__ Qb, const float* __restrict__ Kb,
                 const float* __restrict__ Vb,
                 const float* __restrict__ W_e, const float* __restrict__ W_g,
                 const float* __restrict__ O_e, const float* __restrict__ O_n,
                 float* __restrict__ n_out, float* __restrict__ e_out)
{
    const int row = blockIdx.x;             // b*512 + n
    const int b = row >> 9, nn = row & 511;
    const int tid  = threadIdx.x;
    const int lane = tid & 31;
    const int wv   = tid >> 5;              // wave 0..3
    const int c    = lane & 15;             // WMMA column / A-row-within-16
    const int hi   = lane >> 4;             // half-wave select

    __shared__ float lds_E[NN * LDSW];      // _E[m][h], padded stride, 18 KB
    __shared__ float q_lds[DE];             // Q row for all heads
    __shared__ float gsum[NH];              // sum_m sigmoid(G)
    __shared__ float vbuf[DE];              // centrality-scaled _V, [h*8+k]

    // ---- setup ----
    if (tid < DE) {
        const int h = tid >> 3, kk = tid & 7;
        q_lds[tid] = Qb[(((size_t)(b * NH + h) * NN) + nn) * DK + kk];
    }
    if (tid < NH) gsum[tid] = 0.f;

    // B fragments for projection: [W_e | W_g], K=64 split into two WMMAs
    v16h bw0, bw1;
#pragma unroll
    for (int i = 0; i < 16; ++i) {
        const int kw = ((i < 8) ? i : i + 8) + hi * 8;  // K within 32
        const float w0 = (c < 8) ? W_e[kw * NH + c]        : W_g[kw * NH + (c - 8)];
        const float w1 = (c < 8) ? W_e[(kw + 32) * NH + c] : W_g[(kw + 32) * NH + (c - 8)];
        bw0[i] = (_Float16)w0;
        bw1[i] = (_Float16)w1;
    }
    // B fragments for e_out: O_e [8 x 64], K padded 8 -> 32 with zeros
    v16h bo[4];
#pragma unroll
    for (int t4 = 0; t4 < 4; ++t4) {
#pragma unroll
        for (int i = 0; i < 16; ++i) {
            const int kw = ((i < 8) ? i : i + 8) + hi * 8;
            bo[t4][i] = (_Float16)((kw < 8) ? O_e[kw * DE + t4 * 16 + c] : 0.f);
        }
    }
    __syncthreads();

    const size_t erow_base = ((size_t)b * NN + nn) * NN;   // row base in (m) units
    const float scale = 0.3535533905932738f;               // 1/sqrt(8)

    // ---- phase 1: project e -> E,G ; add clipped QK scores ; fill lds_E ----
    float gpart = 0.f;
    for (int mt = wv; mt < 32; mt += 4) {
        const int m0 = mt * 16;
        const float* erow = e_in + (erow_base + (size_t)(m0 + c)) * DE;
        // prefetch next tile for this lane (global_prefetch_b8 path)
        if (mt + 4 < 32)
            __builtin_prefetch(erow + 4 * 16 * DE, 0, 1);
        v16h a0 = pack_e_frag_nt(erow + hi * 8,      erow + 16 + hi * 8);
        v16h a1 = pack_e_frag_nt(erow + 32 + hi * 8, erow + 48 + hi * 8);
        v8f cc = {};
        cc = __builtin_amdgcn_wmma_f32_16x16x32_f16(false, a0, false, bw0,
                                                    (short)0, cc, false, false);
        cc = __builtin_amdgcn_wmma_f32_16x16x32_f16(false, a1, false, bw1,
                                                    (short)0, cc, false, false);
        if (c < 8) {
            const float* qh = q_lds + c * DK;
#pragma unroll
            for (int r = 0; r < 8; ++r) {
                const int m = m0 + hi * 8 + r;
                const float* kp = Kb + (((size_t)(b * NH + c) * NN) + m) * DK;
                float s = 0.f;
#pragma unroll
                for (int k = 0; k < DK; ++k) s += qh[k] * kp[k];
                s *= scale;
                s = fminf(5.f, fmaxf(-5.f, s));
                lds_E[m * LDSW + c] = cc[r] + s;
            }
        } else {
#pragma unroll
            for (int r = 0; r < 8; ++r)
                gpart += fast_sigmoid(cc[r]);
        }
    }
    if (c >= 8) atomicAdd(&gsum[c - 8], gpart);
    __syncthreads();

    // ---- phase 2: e_out[b,n,m,:] = _E[m,:8] @ O_e   (K padded to 32) ----
    for (int mt = wv; mt < 32; mt += 4) {
        const int m0 = mt * 16;
        v16h aE;
#pragma unroll
        for (int i = 0; i < 16; ++i) aE[i] = (_Float16)0.f;
        if (hi == 0) {                      // lanes 0..15 carry K=0..7 (valid)
            const int m = m0 + c;           // A-matrix row M = lane%16
#pragma unroll
            for (int i = 0; i < 8; ++i) aE[i] = (_Float16)lds_E[m * LDSW + i];
        }
        float* orow = e_out + (erow_base + (size_t)m0) * DE;
#pragma unroll
        for (int t4 = 0; t4 < 4; ++t4) {
            v8f dd = {};
            dd = __builtin_amdgcn_wmma_f32_16x16x32_f16(false, aE, false, bo[t4],
                                                        (short)0, dd, false, false);
#pragma unroll
            for (int r = 0; r < 8; ++r)
                __builtin_nontemporal_store(dd[r],
                    orow + (size_t)(hi * 8 + r) * DE + t4 * 16 + c);
        }
    }

    // ---- phase 3: softmax over m, _V = A @ V, centrality rescale ----
    // wave wv handles heads 2*wv and 2*wv+1  (lds_E/gsum final after barrier)
#pragma unroll
    for (int hh = 0; hh < 2; ++hh) {
        const int h = wv * 2 + hh;
        float mx = -1e30f;
        for (int m = lane; m < NN; m += 32) mx = fmaxf(mx, lds_E[m * LDSW + h]);
#pragma unroll
        for (int o = 16; o > 0; o >>= 1) mx = fmaxf(mx, __shfl_xor(mx, o, 32));
        float se = 0.f;
        float vac[DK];
#pragma unroll
        for (int k = 0; k < DK; ++k) vac[k] = 0.f;
        for (int m = lane; m < NN; m += 32) {
            const float p = __expf(lds_E[m * LDSW + h] - mx);
            se += p;
            const float* vp = Vb + (((size_t)(b * NH + h) * NN) + m) * DK;
#pragma unroll
            for (int k = 0; k < DK; ++k) vac[k] += p * vp[k];
        }
#pragma unroll
        for (int o = 16; o > 0; o >>= 1) {
            se += __shfl_xor(se, o, 32);
#pragma unroll
            for (int k = 0; k < DK; ++k) vac[k] += __shfl_xor(vac[k], o, 32);
        }
        if (lane == 0) {
            const float cent = log1pf(gsum[h]);
            const float sc = cent / se;     // one precise div per head
#pragma unroll
            for (int k = 0; k < DK; ++k) vbuf[h * DK + k] = vac[k] * sc;
        }
    }
    __syncthreads();

    // ---- phase 4: n_out[b,n,:] = vbuf @ O_n ----
    if (tid < DE) {
        float acc = 0.f;
#pragma unroll
        for (int j = 0; j < DE; ++j) acc += vbuf[j] * O_n[j * DE + tid];
        n_out[(size_t)row * DE + tid] = acc;
    }
}

// ------------------------------- launcher ----------------------------------
extern "C" void kernel_launch(void* const* d_in, const int* in_sizes, int n_in,
                              void* d_out, int out_size, void* d_ws, size_t ws_size,
                              hipStream_t stream)
{
    (void)in_sizes; (void)n_in; (void)out_size; (void)ws_size;
    const float* n_ptr  = (const float*)d_in[0];
    const float* e_ptr  = (const float*)d_in[1];
    const float* W_qkv  = (const float*)d_in[2];
    const float* O_n    = (const float*)d_in[3];
    const float* W_g    = (const float*)d_in[4];
    const float* W_e    = (const float*)d_in[5];
    const float* O_e    = (const float*)d_in[6];

    float* out   = (float*)d_out;
    float* n_out = out;                               // [4,512,64]
    float* e_out = out + (size_t)BB * NN * DE;        // [4,512,512,64]

    float* Qb = (float*)d_ws;                         // [B,H,N,dk] each
    float* Kb = Qb + (size_t)BB * NH * NN * DK;       // +131072 floats
    float* Vb = Kb + (size_t)BB * NH * NN * DK;

    qkv_proj_kernel<<<BB * NN, 192, 0, stream>>>(n_ptr, W_qkv, Qb, Kb, Vb);
    attn_edge_kernel<<<BB * NN, 128, 0, stream>>>(e_ptr, Qb, Kb, Vb,
                                                  W_e, W_g, O_e, O_n,
                                                  n_out, e_out);
}
